// MultiHeadAttention_64158221467955
// MI455X (gfx1250) — matrix operand order repaired and checked by reading, not compiled, not run on hardware
//
#include <hip/hip_runtime.h>
#include <hip/hip_bf16.h>

// ---------------------------------------------------------------------------
// Problem constants (match reference)
// ---------------------------------------------------------------------------
#define BB    2
#define LL    2048
#define DDIM  512
#define HH    8
#define KER   3
#define DOUT  512
#define DD    192            // D_DEPTH = KER*DDIM/HH
#define KD    1536           // KER*DDIM
#define SCALE (1.0f / 24.0f) // 1/sqrt(KER*D_DEPTH) = 1/sqrt(576)

typedef __bf16 bf16;
typedef __attribute__((ext_vector_type(16))) __bf16 v16bf;
typedef __attribute__((ext_vector_type(8)))  __bf16 v8bf;
typedef __attribute__((ext_vector_type(8)))  float  v8f;

// ---------------------------------------------------------------------------
// gfx1250 async global->LDS copies (ASYNCcnt path), with sync fallback.
// Builtin signature (from hipcc diagnostic): param0 is
//   'int __attribute__((vector_size(16))) __device__ *'  (addrspace(1))
// so we cast to AS-qualified v4i pointers explicitly.
// ---------------------------------------------------------------------------
#if __has_builtin(__builtin_amdgcn_global_load_async_to_lds_b128) && \
    __has_builtin(__builtin_amdgcn_s_wait_asynccnt)
#define HAVE_ASYNC_LDS 1
#endif

typedef int v4i __attribute__((vector_size(16)));
typedef __attribute__((address_space(1))) v4i* gas_v4i;
typedef __attribute__((address_space(3))) v4i* las_v4i;

static __device__ inline void cp16_g2l(const bf16* g, bf16* l) {
#ifdef HAVE_ASYNC_LDS
  __builtin_amdgcn_global_load_async_to_lds_b128(
      (gas_v4i)(v4i*)g, (las_v4i)(v4i*)l, 0, 0);
#else
  *(v8bf*)l = *(const v8bf*)g;
#endif
}

static __device__ inline void async_fence() {
#ifdef HAVE_ASYNC_LDS
  __builtin_amdgcn_s_wait_asynccnt(0);
#endif
}

// ---------------------------------------------------------------------------
// WMMA helpers (CDNA5 v_wmma_f32_16x16x32_bf16, wave32)
// ---------------------------------------------------------------------------
static __device__ inline v8f wmma_bf16(v16bf a, v16bf b, v8f c) {
  return __builtin_amdgcn_wmma_f32_16x16x32_bf16(false, a, false, b, (short)0, c,
                                                 false, false);
}

static __device__ inline v8f vzero8() {
  v8f z;
#pragma unroll
  for (int i = 0; i < 8; ++i) z[i] = 0.0f;
  return z;
}

// A-operand (16x32 bf16): lane m = lane&15, g = lane>>4.
// elements 0..7 = K (g*8..g*8+7), 8..15 = K (16+g*8..23+g*8). p 16B aligned.
static __device__ inline v16bf loadA16(const bf16* __restrict__ p) {
  v8bf a = *(const v8bf*)p;
  v8bf b = *(const v8bf*)(p + 16);
  v16bf r;
#pragma unroll
  for (int i = 0; i < 8; ++i) { r[i] = a[i]; r[i + 8] = b[i]; }
  return r;
}

// B-operand (32x16 bf16): lane col n = lane&15, g = lane>>4.
// elements 0..7 = K (g*8..g*8+7), 8..15 = K (16+g*8..23+g*8). p 16B aligned.
static __device__ inline v16bf loadB16(const bf16* __restrict__ p) {
  v8bf a = *(const v8bf*)p;
  v8bf b = *(const v8bf*)(p + 16);
  v16bf r;
#pragma unroll
  for (int i = 0; i < 8; ++i) { r[i] = a[i]; r[i + 8] = b[i]; }
  return r;
}

// ---------------------------------------------------------------------------
// fp32 -> bf16 flat convert
// ---------------------------------------------------------------------------
__global__ void cvt_flat_kernel(const float* __restrict__ in,
                                bf16* __restrict__ out, int n) {
  int id = blockIdx.x * blockDim.x + threadIdx.x;
  if (id < n) out[id] = (bf16)in[id];
}

// fp32 [K,N] -> bf16 transposed [N,K]
__global__ void cvt_T_kernel(const float* __restrict__ in,
                             bf16* __restrict__ out, int K, int N) {
  int id = blockIdx.x * blockDim.x + threadIdx.x;
  if (id >= N * K) return;
  int k = id % K;
  int n = id / K;
  out[id] = (bf16)in[(size_t)k * N + n];
}

// ---------------------------------------------------------------------------
// GEMM: C[M,N] = A[M,K](bf16) * BT[N,K](bf16)^T + bias.
// Block 256 = 8 waves (4x2); each wave computes 16x32; K stepped by 32.
// ---------------------------------------------------------------------------
__global__ __launch_bounds__(256)
void gemm_bf16_kernel(const bf16* __restrict__ A, const bf16* __restrict__ BT,
                      const float* __restrict__ bias,
                      float* __restrict__ Cf, bf16* __restrict__ Cb,
                      int M, int N, int K, int out_bf16) {
  const int tid  = threadIdx.x;
  const int wave = tid >> 5;
  const int lane = tid & 31;
  const int g    = lane >> 4;
  const int c    = lane & 15;
  const int wrow = wave >> 1;
  const int wcol = wave & 1;

  const int row0 = blockIdx.x * 64 + wrow * 16;
  const int n0   = blockIdx.y * 64 + wcol * 32;

  v8f acc0 = vzero8(), acc1 = vzero8();

  const bf16* Arow  = A + (size_t)(row0 + c) * K;
  const bf16* Brow0 = BT + (size_t)(n0 + c) * K;
  const bf16* Brow1 = BT + (size_t)(n0 + 16 + c) * K;

  for (int kk = 0; kk < K; kk += 32) {
    v16bf a  = loadA16(Arow + kk + g * 8);
    v16bf b0 = loadB16(Brow0 + kk + g * 8);
    v16bf b1 = loadB16(Brow1 + kk + g * 8);
    acc0 = wmma_bf16(a, b0, acc0);
    acc1 = wmma_bf16(a, b1, acc1);
  }

#pragma unroll
  for (int r = 0; r < 8; ++r) {
    int row = row0 + r + 8 * g;
    float v0 = acc0[r] + bias[n0 + c];
    float v1 = acc1[r] + bias[n0 + 16 + c];
    size_t o0 = (size_t)row * N + n0 + c;
    size_t o1 = (size_t)row * N + n0 + 16 + c;
    if (out_bf16) { Cb[o0] = (bf16)v0; Cb[o1] = (bf16)v1; }
    else          { Cf[o0] = v0;       Cf[o1] = v1; }
  }
}

// ---------------------------------------------------------------------------
// Patch/head gather.
//  mode 0: pq -> Qp  [BH, L, DD]
//  mode 1: pk -> Kp  [BH, L, DD]
//  mode 2: pv -> VpT [BH, DD, L]  (transposed: attn*V B-loads contiguous)
// ---------------------------------------------------------------------------
__global__ void patch_kernel(const bf16* __restrict__ pq,
                             const bf16* __restrict__ pk,
                             const bf16* __restrict__ pv,
                             bf16* __restrict__ Qp, bf16* __restrict__ Kp,
                             bf16* __restrict__ VpT) {
  const int mode = blockIdx.y;
  const int id = blockIdx.x * blockDim.x + threadIdx.x;
  int b, h, t, d;
  if (mode < 2) {
    d = id % DD;
    int rest = id / DD;
    t = rest % LL;
    int bh = rest / LL;
    b = bh / HH; h = bh % HH;
  } else {
    t = id % LL;
    int rest = id / LL;
    d = rest % DD;
    int bh = rest / DD;
    b = bh / HH; h = bh % HH;
  }
  const int f  = h * DD + d;
  const int i  = f / DDIM;
  const int ch = f % DDIM;
  const int st = t + i - (KER - 1);
  const bf16* src = (mode == 0) ? pq : (mode == 1) ? pk : pv;
  bf16 val = (bf16)0.0f;
  if (st >= 0) val = src[((size_t)b * LL + st) * DDIM + ch];
  if (mode == 0)      Qp[id]  = val;
  else if (mode == 1) Kp[id]  = val;
  else                VpT[id] = val;
}

// ---------------------------------------------------------------------------
// Fused attention.  Grid (L/64, B*H), block 128 = 4 waves, 16 queries/wave.
// K (and V in pass 2) panels of 32 keys are staged cooperatively into LDS
// with gfx1250 async global->LDS b128 copies, double buffered: each chunk
// issues the next chunk's DMA right after the barrier so it overlaps the
// WMMA + exp work of the current chunk.
// ---------------------------------------------------------------------------
__global__ __launch_bounds__(128)
void attn_kernel(const bf16* __restrict__ Qp, const bf16* __restrict__ Kp,
                 const bf16* __restrict__ VpT,
                 float* __restrict__ attn, bf16* __restrict__ ctx) {
  __shared__ alignas(16) bf16 sK[2][32 * DD];   // 2 x 12 KB  (key-major rows)
  __shared__ alignas(16) bf16 sV[2][DD * 32];   // 2 x 12 KB  (d-major rows)
  __shared__ alignas(16) bf16 sP[4][16 * 32];   // per-wave P staging

  const int tid  = threadIdx.x;
  const int wave = tid >> 5;
  const int lane = tid & 31;
  const int g    = lane >> 4;
  const int c    = lane & 15;
  const int bh   = blockIdx.y;
  const int b    = bh >> 3;
  const int h    = bh & 7;
  const int q0   = blockIdx.x * 64 + wave * 16;

  const bf16* Kbh = Kp + (size_t)bh * LL * DD;
  const bf16* Vbh = VpT + (size_t)bh * DD * LL;

  // Cooperative stagers: 768 b128 segments each, 6 per thread, coalesced.
  auto stageK = [&](int kc, int buf) {
#pragma unroll
    for (int i = 0; i < 6; ++i) {
      int seg = i * 128 + tid;        // 0..767
      int key = seg / 24;             // 32 keys
      int off = (seg % 24) * 8;       // 192 d, 8 bf16 per b128
      cp16_g2l(Kbh + (size_t)(kc * 32 + key) * DD + off,
               &sK[buf][key * DD + off]);
    }
  };
  auto stageV = [&](int kc, int buf) {
#pragma unroll
    for (int i = 0; i < 6; ++i) {
      int seg = i * 128 + tid;        // 0..767
      int d   = seg >> 2;             // 192 rows
      int ko  = (seg & 3) * 8;        // 32 keys, 8 bf16 per b128
      cp16_g2l(Vbh + (size_t)d * LL + kc * 32 + ko,
               &sV[buf][d * 32 + ko]);
    }
  };

  // Preload 6 Q A-tiles (16 x 192), kept in VGPRs for both passes.
  const bf16* qrow = Qp + ((size_t)bh * LL + q0 + c) * DD;
  v16bf qa[6];
#pragma unroll
  for (int j = 0; j < 6; ++j) qa[j] = loadA16(qrow + j * 32 + g * 8);

  float m[8], s[8];
#pragma unroll
  for (int r = 0; r < 8; ++r) { m[r] = -1e30f; s[r] = 0.0f; }

  const int NCHUNK = LL / 32;  // 64 chunks of 32 keys

  // ---- Pass 1: row max / exp-sum (K panels only) ----
  stageK(0, 0);
  for (int kc = 0; kc < NCHUNK; ++kc) {
    const int buf = kc & 1;
    async_fence();
    __syncthreads();
    if (kc + 1 < NCHUNK) stageK(kc + 1, buf ^ 1);
#pragma unroll
    for (int half = 0; half < 2; ++half) {
      v8f S = vzero8();
      const bf16* krow = &sK[buf][(half * 16 + c) * DD];
#pragma unroll
      for (int j = 0; j < 6; ++j)
        S = wmma_bf16(qa[j], loadB16(krow + j * 32 + g * 8), S);
#pragma unroll
      for (int r = 0; r < 8; ++r) {
        float x  = S[r] * SCALE;
        float nm = fmaxf(m[r], x);
        s[r] = s[r] * __expf(m[r] - nm) + __expf(x - nm);
        m[r] = nm;
      }
    }
  }

  // Log-sum-exp butterfly across the 16-lane row group (wave32 halves).
#pragma unroll
  for (int r = 0; r < 8; ++r) {
#pragma unroll
    for (int mask = 1; mask < 16; mask <<= 1) {
      float mo = __shfl_xor(m[r], mask, 32);
      float so = __shfl_xor(s[r], mask, 32);
      float nm = fmaxf(m[r], mo);
      s[r] = s[r] * __expf(m[r] - nm) + so * __expf(mo - nm);
      m[r] = nm;
    }
  }
  float inv[8];
#pragma unroll
  for (int r = 0; r < 8; ++r) inv[r] = 1.0f / s[r];

  // ---- Pass 2: softmax write + ctx accumulate (K and V panels) ----
  v8f acc[12];
#pragma unroll
  for (int n = 0; n < 12; ++n) acc[n] = vzero8();

  bf16* sPw = sP[wave];
  float* attnBase = attn + ((size_t)bh * LL + q0) * LL;

  __syncthreads();
  stageK(0, 0);
  stageV(0, 0);
  for (int kc = 0; kc < NCHUNK; ++kc) {
    const int buf = kc & 1;
    async_fence();
    __syncthreads();
    if (kc + 1 < NCHUNK) { stageK(kc + 1, buf ^ 1); stageV(kc + 1, buf ^ 1); }
#pragma unroll
    for (int half = 0; half < 2; ++half) {
      const int kt = kc * 2 + half;
      v8f S = vzero8();
      const bf16* krow = &sK[buf][(half * 16 + c) * DD];
#pragma unroll
      for (int j = 0; j < 6; ++j)
        S = wmma_bf16(qa[j], loadB16(krow + j * 32 + g * 8), S);
#pragma unroll
      for (int r = 0; r < 8; ++r) {
        float p = __expf(S[r] * SCALE - m[r]) * inv[r];
        attnBase[(size_t)(r + 8 * g) * LL + kt * 16 + c] = p;  // exact fp32 attn
        sPw[(r + 8 * g) * 32 + half * 16 + c] = (bf16)p;       // C/D -> A staging
      }
    }
    // Same-wave LDS ops are in-order; sP region is wave-private.
    v16bf pa = loadA16(sPw + c * 32 + g * 8);  // P as 16x32 A-tile
#pragma unroll
    for (int n = 0; n < 12; ++n) {
      v16bf vb = loadB16(&sV[buf][(n * 16 + c) * 32 + g * 8]);
      acc[n] = wmma_bf16(pa, vb, acc[n]);
    }
  }

  // combine_heads: ctx[b, t, h*DD + d]  (row-major [B, L, KD], bf16)
  bf16* ctxBase = ctx + ((size_t)b * LL + q0) * KD + h * DD;
#pragma unroll
  for (int n = 0; n < 12; ++n) {
#pragma unroll
    for (int r = 0; r < 8; ++r)
      ctxBase[(size_t)(r + 8 * g) * KD + n * 16 + c] = (bf16)acc[n][r];
  }
}

// ---------------------------------------------------------------------------
// Host launcher
// ---------------------------------------------------------------------------
extern "C" void kernel_launch(void* const* d_in, const int* in_sizes, int n_in,
                              void* d_out, int out_size, void* d_ws,
                              size_t ws_size, hipStream_t stream) {
  const float* q  = (const float*)d_in[0];
  const float* k  = (const float*)d_in[1];
  const float* v  = (const float*)d_in[2];
  const float* wq = (const float*)d_in[3];
  const float* bq = (const float*)d_in[4];
  const float* wk = (const float*)d_in[5];
  const float* bk = (const float*)d_in[6];
  const float* wv = (const float*)d_in[7];
  const float* bv = (const float*)d_in[8];
  const float* wo = (const float*)d_in[9];
  const float* bo = (const float*)d_in[10];

  float* out  = (float*)d_out;                       // [B, L, DOUT]
  float* attn = out + (size_t)BB * LL * DOUT;        // [B, H, L, L]

  char* ws = (char*)d_ws;
  const size_t SZ_IN  = (size_t)BB * LL * DDIM * sizeof(bf16);
  const size_t SZ_W   = (size_t)DDIM * DDIM * sizeof(bf16);
  const size_t SZ_WO  = (size_t)DOUT * KD * sizeof(bf16);
  const size_t SZ_HP  = (size_t)BB * HH * LL * DD * sizeof(bf16);
  const size_t SZ_CTX = (size_t)BB * LL * KD * sizeof(bf16);

  size_t off = 0;
  bf16* qb  = (bf16*)(ws + off); off += SZ_IN;
  bf16* kb  = (bf16*)(ws + off); off += SZ_IN;
  bf16* vb  = (bf16*)(ws + off); off += SZ_IN;
  bf16* wqT = (bf16*)(ws + off); off += SZ_W;
  bf16* wkT = (bf16*)(ws + off); off += SZ_W;
  bf16* wvT = (bf16*)(ws + off); off += SZ_W;
  bf16* woT = (bf16*)(ws + off); off += SZ_WO;
  bf16* pq  = (bf16*)(ws + off); off += SZ_IN;
  bf16* pk  = (bf16*)(ws + off); off += SZ_IN;
  bf16* pv  = (bf16*)(ws + off); off += SZ_IN;
  bf16* Qp  = (bf16*)(ws + off); off += SZ_HP;
  bf16* Kp  = (bf16*)(ws + off); off += SZ_HP;
  bf16* VpT = (bf16*)(ws + off); off += SZ_HP;
  bf16* ctx = (bf16*)(ws + off); off += SZ_CTX;
  (void)ws_size; (void)in_sizes; (void)n_in; (void)out_size;

  const int M = BB * LL;        // 4096
  const int nflat = M * DDIM;   // 2,097,152

  cvt_flat_kernel<<<nflat / 256, 256, 0, stream>>>(q, qb, nflat);
  cvt_flat_kernel<<<nflat / 256, 256, 0, stream>>>(k, kb, nflat);
  cvt_flat_kernel<<<nflat / 256, 256, 0, stream>>>(v, vb, nflat);

  cvt_T_kernel<<<(DDIM * DDIM) / 256, 256, 0, stream>>>(wq, wqT, DDIM, DDIM);
  cvt_T_kernel<<<(DDIM * DDIM) / 256, 256, 0, stream>>>(wk, wkT, DDIM, DDIM);
  cvt_T_kernel<<<(DDIM * DDIM) / 256, 256, 0, stream>>>(wv, wvT, DDIM, DDIM);
  cvt_T_kernel<<<(DOUT * KD) / 256, 256, 0, stream>>>(wo, woT, KD, DOUT);

  dim3 ggrid(M / 64, DDIM / 64);
  gemm_bf16_kernel<<<ggrid, 256, 0, stream>>>(qb, wqT, bq, nullptr, pq,
                                              M, DDIM, DDIM, 1);
  gemm_bf16_kernel<<<ggrid, 256, 0, stream>>>(kb, wkT, bk, nullptr, pk,
                                              M, DDIM, DDIM, 1);
  gemm_bf16_kernel<<<ggrid, 256, 0, stream>>>(vb, wvT, bv, nullptr, pv,
                                              M, DDIM, DDIM, 1);

  const int ngather = BB * HH * LL * DD;
  patch_kernel<<<dim3(ngather / 256, 3), 256, 0, stream>>>(pq, pk, pv,
                                                           Qp, Kp, VpT);

  attn_kernel<<<dim3(LL / 64, BB * HH), 128, 0, stream>>>(Qp, Kp, VpT,
                                                          attn, ctx);

  dim3 ogrid(M / 64, DOUT / 64);
  gemm_bf16_kernel<<<ogrid, 256, 0, stream>>>(ctx, woT, bo, out, nullptr,
                                              M, DOUT, KD, 0);
}
